// MSARowAttentionWithPairBias_18528488915225
// MI455X (gfx1250) — compile-verified, matrix operand order checked by reading, not run on hardware
//
#include <hip/hip_runtime.h>

// MSARowAttentionWithPairBias on MI455X (gfx1250, wave32, WMMA bf16).
// S=128 R=256 CM=256 CZ=128 C=32 H=8.  ~47 GFLOP -> matrix-core bound;
// all GEMMs run on v_wmma_f32_16x16x32_bf16 with fp32 accumulation.
// Attention operand stage uses the Tensor Data Mover (tensor_load_to_lds)
// when the toolchain exposes it. Workspace use: ~253 MB.

typedef unsigned short u16;
typedef __attribute__((ext_vector_type(16))) __bf16 v16bf;
typedef __attribute__((ext_vector_type(8)))  float  v8f;
typedef __attribute__((ext_vector_type(4)))  unsigned int u32x4;
typedef __attribute__((ext_vector_type(8)))  int i32x8;
typedef __attribute__((ext_vector_type(4)))  int i32x4;

__device__ __forceinline__ u16 f2bf(float f) {
  unsigned u = __float_as_uint(f);
  u += 0x7fffu + ((u >> 16) & 1u);
  return (u16)(u >> 16);
}
__device__ __forceinline__ float bf2f(u16 h) {
  return __uint_as_float(((unsigned)h) << 16);
}

// Load one 16-lane-striped bf16 WMMA fragment from LDS.
// ISA 16-bit layout: lane-local packed K pairs at [kbase..kbase+7] and
// [kbase+16..kbase+23] -> two ds_load_b128.
__device__ __forceinline__ v16bf ld_frag(const u16* p) {
  union { uint4 q[2]; v16bf v; } u;
  u.q[0] = *(const uint4*)(p);
  u.q[1] = *(const uint4*)(p + 16);
  return u.v;
}

__device__ __forceinline__ v8f wmma_bf(v16bf a, v16bf b, v8f c) {
  return __builtin_amdgcn_wmma_f32_16x16x32_bf16(false, a, false, b, (short)0, c,
                                                 false, false);
}

#if __has_builtin(__builtin_amdgcn_tensor_load_to_lds)
#define HAVE_TDM 1
#pragma message("CDNA5 probe: tensor_load_to_lds builtin AVAILABLE -> TDM path compiled")
// One-shot contiguous TDM load: nelem bf16 elements (tile_dim0 = tensor_dim0),
// global -> LDS, tracked on TENSORcnt.  D# per CDNA5 ISA ch.8.
__device__ __forceinline__ void tdm_load_contig(unsigned lds_addr,
                                                unsigned long long gaddr,
                                                unsigned nelem) {
  u32x4 g0;
  g0.x = 1u;                                        // count=1, user desc
  g0.y = lds_addr;                                  // LDS byte address
  g0.z = (unsigned)(gaddr & 0xffffffffull);         // global_addr[31:0]
  g0.w = (unsigned)((gaddr >> 32) & 0x1ffffffull)   // global_addr[56:32]
         | (2u << 30);                              // type=2 ("image")
  i32x8 g1;
  g1[0] = (1 << 16);                  // data_size=1 -> 2-byte elements
  g1[1] = (int)(nelem << 16);         // tensor_dim0[15:0] @ bits 63:48
  g1[2] = (int)((nelem >> 16) | (1u << 16)); // tensor_dim0[31:16]; tensor_dim1=1
  g1[3] = (int)((nelem & 0xffffu) << 16);    // tile_dim0 @ bits 127:112
  g1[4] = 1;                          // tile_dim1=1, tile_dim2=0
  g1[5] = (int)nelem;                 // tensor_dim0_stride[31:0]
  g1[6] = (int)((nelem & 0xffffu) << 16);    // tensor_dim1_stride[15:0]
  g1[7] = 0;
  i32x4 gz = {0, 0, 0, 0};
#if defined(__clang_major__) && __clang_major__ >= 23
  i32x8 gz8 = {0, 0, 0, 0, 0, 0, 0, 0};
  __builtin_amdgcn_tensor_load_to_lds(g0, g1, gz, gz, gz8, 0);
#else
  __builtin_amdgcn_tensor_load_to_lds(g0, g1, gz, gz, 0);
#endif
}
#else
#define HAVE_TDM 0
#pragma message("CDNA5 probe: tensor_load_to_lds builtin MISSING -> fallback copy path")
#endif

#if defined(__clang_major__)
#if __clang_major__ >= 23
#pragma message("CDNA5 probe: clang major >= 23 (6-arg TDM form)")
#else
#pragma message("CDNA5 probe: clang major <= 22 (5-arg TDM form)")
#endif
#endif

// ---------------- LayerNorm (no affine; affine folded into weights) --------
// One wave per row; rowlen in {128, 256}; block = 256 threads = 8 rows.
__global__ void k_layernorm_rows(const float* __restrict__ x,
                                 u16* __restrict__ out, int rowlen) {
  int row  = blockIdx.x * 8 + (threadIdx.x >> 5);
  int lane = threadIdx.x & 31;
  int per  = rowlen >> 5;
  size_t base = (size_t)row * rowlen;
  float v[8];
  float s = 0.f, ss = 0.f;
  for (int e = 0; e < per; ++e) {
    float t = x[base + lane + e * 32];
    v[e] = t; s += t; ss += t * t;
  }
  for (int off = 16; off > 0; off >>= 1) {
    s  += __shfl_xor(s, off);
    ss += __shfl_xor(ss, off);
  }
  float mean = s / rowlen;
  float var  = ss / rowlen - mean * mean;
  float inv  = rsqrtf(var + 1e-5f);
  for (int e = 0; e < per; ++e)
    out[base + lane + e * 32] = f2bf((v[e] - mean) * inv);
}

// ---------------- Weight folding ------------------------------------------
// Wcat[h, c, n] (n: 0..31 q | 32..63 k | 64..95 v | 96..127 g), gamma folded.
__global__ void k_fold_qkvg(const float* __restrict__ Wq, const float* __restrict__ Wk,
                            const float* __restrict__ Wv, const float* __restrict__ Wg,
                            const float* __restrict__ gam, u16* __restrict__ Wcat) {
  int b = blockIdx.x;            // h*256 + c
  int n = threadIdx.x, sel = n >> 5, d = n & 31;
  const float* W = (sel == 0) ? Wq : (sel == 1) ? Wk : (sel == 2) ? Wv : Wg;
  Wcat[(size_t)b * 128 + n] = f2bf(gam[b] * W[(size_t)b * 32 + d]);
}

__global__ void k_fold_bc(const float* __restrict__ Wq, const float* __restrict__ Wk,
                          const float* __restrict__ Wv, const float* __restrict__ Wg,
                          const float* __restrict__ beta, float* __restrict__ bc) {
  int h = blockIdx.x;
  int n = threadIdx.x, sel = n >> 5, d = n & 31;
  const float* W = (sel == 0) ? Wq : (sel == 1) ? Wk : (sel == 2) ? Wv : Wg;
  float sum = 0.f;
  for (int c = 0; c < 256; ++c)
    sum += beta[h * 256 + c] * W[(size_t)(h * 256 + c) * 32 + d];
  bc[h * 128 + n] = sum;
}

__global__ void k_fold_wb(const float* __restrict__ Wb, const float* __restrict__ gam,
                          u16* __restrict__ Wbf) {
  int b = blockIdx.x;            // h*128 + c
  int t = threadIdx.x;           // s2
  Wbf[(size_t)b * 128 + t] = f2bf(gam[b] * Wb[(size_t)b * 128 + t]);
}

__global__ void k_fold_bb(const float* __restrict__ Wb, const float* __restrict__ beta,
                          float* __restrict__ bb) {
  int h = blockIdx.x, t = threadIdx.x;
  float sum = 0.f;
  for (int c = 0; c < 128; ++c)
    sum += beta[h * 128 + c] * Wb[(size_t)(h * 128 + c) * 128 + t];
  bb[h * 128 + t] = sum;
}

__global__ void k_cvt_bf16(const float* __restrict__ x, u16* __restrict__ y) {
  int i = blockIdx.x * 256 + threadIdx.x;
  y[i] = f2bf(x[i]);
}

// ---------------- Generic bf16 WMMA GEMM ----------------------------------
// C[m,n] = sum_k A[m,k]*B[k,n] + bias[n].  Tile 128x128, block 256 (8 waves),
// wave w owns M-rows [w*16, w*16+16) x 8 N-subtiles.  B staged N-major in LDS.
__global__ void k_gemm_bf16(const u16* __restrict__ A, const u16* __restrict__ B,
                            const float* __restrict__ bias, void* __restrict__ Cout,
                            int K, int ldB, int ldC,
                            long sA, long sB, long sC, long sBias, int outF32) {
  int z = blockIdx.z;
  A    += (size_t)z * sA;
  B    += (size_t)z * sB;
  bias += (size_t)z * sBias;
  int mtile = blockIdx.x;
  int n0    = blockIdx.y * 128;

  extern __shared__ u16 sm[];
  u16* Bt  = sm;            // 128 (n) x K, row-contiguous in k
  u16* Ast = sm + 128 * K;  // 128 (m) x 32

  int tid = threadIdx.x;
  {  // K is a power of two (128 or 256): shift/mask instead of div/mod
    int ksh = __builtin_ctz(K);
    int kmask = K - 1;
    for (int idx = tid; idx < 128 * K; idx += 256) {
      int n = idx >> ksh, k = idx & kmask;
      Bt[idx] = B[(size_t)k * ldB + n0 + n];
    }
  }
  __syncthreads();

  int lane = tid & 31, w = tid >> 5;
  int lidx = lane & 15, kbase = (lane >> 4) * 8;

  v8f acc[8];
  for (int nt = 0; nt < 8; ++nt)
    for (int e = 0; e < 8; ++e) acc[nt][e] = 0.f;

  int nsteps = K >> 5;
  for (int kb = 0; kb < nsteps; ++kb) {
    {  // stage A 128x32: each thread copies 16 bf16 (32B)
      int row = tid >> 1, half = tid & 1;
      const u16* src = A + ((size_t)(mtile * 128 + row)) * K + kb * 32 + half * 16;
      uint4* dst = (uint4*)(Ast + row * 32 + half * 16);
      dst[0] = ((const uint4*)src)[0];
      dst[1] = ((const uint4*)src)[1];
      if (kb + 1 < nsteps) __builtin_prefetch(src + 32, 0, 1);  // next K panel
    }
    __syncthreads();
    v16bf af = ld_frag(Ast + (w * 16 + lidx) * 32 + kbase);
    for (int nt = 0; nt < 8; ++nt) {
      v16bf bfr = ld_frag(Bt + (size_t)(nt * 16 + lidx) * K + kb * 32 + kbase);
      acc[nt] = wmma_bf(af, bfr, acc[nt]);
    }
    __syncthreads();
  }

  for (int nt = 0; nt < 8; ++nt)
    for (int r = 0; r < 8; ++r) {
      int m = mtile * 128 + w * 16 + ((lane >> 4) << 3) + r;
      int n = n0 + nt * 16 + lidx;
      float val = acc[nt][r] + bias[n];
      size_t off = (size_t)z * sC + (size_t)m * ldC + n;
      if (outF32) ((float*)Cout)[off] = val;
      else        ((u16*)Cout)[off]   = f2bf(val);
    }
}

// ---------------- Fused attention per (h, s) ------------------------------
// scores = (q k^T + b)/sqrt(C); softmax over the QUERY axis i (axis=-2),
// processed in two j-halves of 128 columns (score half = 128KB fp32 in LDS);
// O accumulators stay in VGPRs across halves; sigmoid(g) gating at the end.
// The whole q|k|v|g block for (h,s) is one contiguous 64KB slab of qkvg and
// is staged into LDS by the Tensor Data Mover when available.
__global__ void k_attention(const u16* __restrict__ qkvg,
                            const u16* __restrict__ bflat,
                            u16* __restrict__ obuf) {
  int h = blockIdx.x >> 7;
  int s = blockIdx.x & 127;

  extern __shared__ u16 sm[];
  u16*   stage = sm;                      // 256x128 bf16 (64KB): q|k|v|g rows
  u16*   vt_s  = sm + 32768;              // 32x256 bf16  (16KB): v transposed
  float* sc    = (float*)(sm + 40960);    // 256x128 f32  (128KB): score half
  u16*   a_s   = (u16*)(sc + 256 * 128);  // 256x128 bf16 (64KB): softmaxed A

  int tid = threadIdx.x;
  const u16* blk = qkvg + ((size_t)h * 32768 + (size_t)s * 256) * 128;

#if HAVE_TDM
  if (tid < 32) {  // wave 0 issues one TDM descriptor for the 64KB block
    unsigned lds_addr = (unsigned)(size_t)(void*)stage;
    lds_addr = __builtin_amdgcn_readfirstlane(lds_addr);
    tdm_load_contig(lds_addr, (unsigned long long)(size_t)blk, 32768u);
    __builtin_amdgcn_s_wait_tensorcnt(0);
  }
#else
  {  // fallback: cooperative copy, first 128B (q|k cols) of one row per thread
    uint4* dst = (uint4*)(stage + tid * 128);
    const uint4* src = (const uint4*)(blk + (size_t)tid * 128);
    for (int c = 0; c < 8; ++c) dst[c] = src[c];
  }
#endif
  __syncthreads();

  {  // build v^T (32 x 256) from the staged block
    u16 vtmp[32];
    const uint4* vsrc = (const uint4*)(stage + tid * 128 + 64);
    for (int c = 0; c < 4; ++c) ((uint4*)vtmp)[c] = vsrc[c];
    for (int d = 0; d < 32; ++d) vt_s[d * 256 + tid] = vtmp[d];
  }
  __syncthreads();

  int lane = tid & 31, w = tid >> 5;
  int lidx = lane & 15, lh8 = (lane >> 4) * 8, kbase = lh8;

  v8f oacc[2][2];
  for (int a = 0; a < 2; ++a)
    for (int b = 0; b < 2; ++b)
      for (int e = 0; e < 8; ++e) oacc[a][b][e] = 0.f;

  const float isc = 0.17677669529663687f;  // 1/sqrt(32)

  for (int jh = 0; jh < 2; ++jh) {
    // ---- scores: q(256x32) @ k^T, +bias, scaled, into LDS fp32 ----
    for (int mt = 0; mt < 2; ++mt) {
      int mbase = w * 32 + mt * 16;
      v16bf af = ld_frag(stage + (mbase + lidx) * 128 + kbase);       // q cols 0..31
      for (int nt = 0; nt < 8; ++nt) {
        int j = jh * 128 + nt * 16 + lidx;
        v16bf bfr = ld_frag(stage + j * 128 + 32 + kbase);            // k cols 32..63
        v8f p;
        for (int e = 0; e < 8; ++e) p[e] = 0.f;
        p = wmma_bf(af, bfr, p);
        size_t bb = (size_t)h * 8388608ull + (size_t)s * 65536ull + (size_t)j;
        for (int r = 0; r < 8; ++r) {
          int i = mbase + lh8 + r;
          float bv = bf2f(bflat[bb + (size_t)i * 256]);
          sc[i * 128 + nt * 16 + lidx] = (p[r] + bv) * isc;
        }
      }
    }
    __syncthreads();

    // ---- column softmax (over i), one thread per column ----
    if (tid < 128) {
      int col = tid;
      float mx = -3.0e38f;
      for (int i = 0; i < 256; ++i) mx = fmaxf(mx, sc[i * 128 + col]);
      float sum = 0.f;
      for (int i = 0; i < 256; ++i) {
        float e = __expf(sc[i * 128 + col] - mx);
        sc[i * 128 + col] = e;
        sum += e;
      }
      float inv = 1.0f / sum;
      for (int i = 0; i < 256; ++i)
        a_s[i * 128 + col] = f2bf(sc[i * 128 + col] * inv);
    }
    __syncthreads();

    // ---- O += A(256x128) @ V(128x32) ----
    for (int kb = 0; kb < 4; ++kb)
      for (int mi = 0; mi < 2; ++mi) {
        int m = w * 32 + mi * 16 + lidx;
        v16bf af = ld_frag(a_s + m * 128 + kb * 32 + kbase);
        for (int di = 0; di < 2; ++di) {
          int n = di * 16 + lidx;
          v16bf bfr = ld_frag(vt_s + n * 256 + jh * 128 + kb * 32 + kbase);
          oacc[mi][di] = wmma_bf(af, bfr, oacc[mi][di]);
        }
      }
    __syncthreads();
  }

  // ---- sigmoid gate (g = staged cols 96..127) + store ----
  for (int mi = 0; mi < 2; ++mi)
    for (int di = 0; di < 2; ++di)
      for (int r = 0; r < 8; ++r) {
        int i = w * 32 + mi * 16 + lh8 + r;
        int d = di * 16 + lidx;
        float g  = bf2f(stage[i * 128 + 96 + d]);
        float sg = 1.0f / (1.0f + __expf(-g));
        obuf[((size_t)(s * 256 + i)) * 256 + h * 32 + d] = f2bf(oacc[mi][di][r] * sg);
      }
}

// ---------------- Host launch ---------------------------------------------
extern "C" void kernel_launch(void* const* d_in, const int* in_sizes, int n_in,
                              void* d_out, int out_size, void* d_ws, size_t ws_size,
                              hipStream_t stream) {
  const float* msa       = (const float*)d_in[0];
  const float* pair      = (const float*)d_in[1];
  const float* ln_msa_g  = (const float*)d_in[2];
  const float* ln_msa_b  = (const float*)d_in[3];
  const float* ln_pair_g = (const float*)d_in[4];
  const float* ln_pair_b = (const float*)d_in[5];
  const float* Wq        = (const float*)d_in[6];
  const float* Wk        = (const float*)d_in[7];
  const float* Wv        = (const float*)d_in[8];
  const float* Wg        = (const float*)d_in[9];
  const float* Wb        = (const float*)d_in[10];
  const float* Wo        = (const float*)d_in[11];
  const float* bo        = (const float*)d_in[12];

  u16* ws    = (u16*)d_ws;
  u16* nm    = ws;                    //  8,388,608 bf16  normalized msa
  u16* npair = nm    + 8388608;       //  8,388,608 bf16  normalized pair
  u16* qkvg  = npair + 8388608;       // 33,554,432 bf16  (H, S*R, 128)
  u16* bfl   = qkvg  + 33554432;      // 67,108,864 bf16  (H, R*R, S)
  u16* obuf  = bfl   + 67108864;      //  8,388,608 bf16  (S, R, H*C)
  u16* Wcat  = obuf  + 8388608;       //    262,144 bf16  folded qkvg weights
  u16* Wbf   = Wcat  + 262144;        //    131,072 bf16  folded pair weights
  u16* Wob   = Wbf   + 131072;        //     65,536 bf16  Wo
  float* bc  = (float*)(Wob + 65536); //      1,024 f32
  float* bb  = bc + 1024;             //      1,024 f32

  size_t lds_gemm_k256 = 8192 + 128 * 256 * 2;  // 73,728 B
  size_t lds_gemm_k128 = 8192 + 128 * 128 * 2;  // 40,960 B
  size_t lds_attn      = 278528;                // 272 KB (of 320 KB/WGP)
  (void)hipFuncSetAttribute((const void*)k_gemm_bf16,
                            hipFuncAttributeMaxDynamicSharedMemorySize,
                            (int)lds_gemm_k256);
  (void)hipFuncSetAttribute((const void*)k_attention,
                            hipFuncAttributeMaxDynamicSharedMemorySize,
                            (int)lds_attn);

  // LayerNorms (affine folded into weights below)
  k_layernorm_rows<<<4096, 256, 0, stream>>>(msa, nm, 256);
  k_layernorm_rows<<<8192, 256, 0, stream>>>(pair, npair, 128);

  // Fold per-head LN affine into projection weights + biases
  k_fold_qkvg<<<2048, 128, 0, stream>>>(Wq, Wk, Wv, Wg, ln_msa_g, Wcat);
  k_fold_bc  <<<   8, 128, 0, stream>>>(Wq, Wk, Wv, Wg, ln_msa_b, bc);
  k_fold_wb  <<<1024, 128, 0, stream>>>(Wb, ln_pair_g, Wbf);
  k_fold_bb  <<<   8, 128, 0, stream>>>(Wb, ln_pair_b, bb);
  k_cvt_bf16 <<< 256, 256, 0, stream>>>(Wo, Wob);

  // QKVG projection: per head (32768 x 256) @ (256 x 128)
  k_gemm_bf16<<<dim3(256, 1, 8), 256, lds_gemm_k256, stream>>>(
      nm, Wcat, bc, qkvg, 256, 128, 128, 0L, 32768L, 4194304L, 128L, 0);

  // Pair bias tensor: per head (65536 x 128) @ (128 x 128)
  k_gemm_bf16<<<dim3(512, 1, 8), 256, lds_gemm_k128, stream>>>(
      npair, Wbf, bb, bfl, 128, 128, 128, 0L, 16384L, 8388608L, 128L, 0);

  // Fused attention per (h, s)
  k_attention<<<1024, 256, lds_attn, stream>>>(qkvg, bfl, obuf);

  // Output projection: (32768 x 256) @ (256 x 256) + bo -> f32 d_out
  k_gemm_bf16<<<dim3(256, 2, 1), 256, lds_gemm_k256, stream>>>(
      obuf, Wob, bo, d_out, 256, 256, 256, 0L, 0L, 0L, 0L, 1);
}
